// MultiHeadLinearAttention_9431748182265
// MI455X (gfx1250) — matrix operand order repaired and checked by reading, not compiled, not run on hardware
//
#include <hip/hip_runtime.h>
#include <hip/hip_bf16.h>
#include <math.h>

// ---------------------------------------------------------------------------
// MultiHeadLinearAttention for MI455X (gfx1250, wave32, WMMA bf16->f32)
//
// Shapes: B=8, S=2048, E=2048, H=16, d=128.
// GEMM core: one wave computes a 32x64 tile via v_wmma_f32_16x16x32_bf16
// (MT=2 M-subtiles x 4 N-subtiles = 8 f32 accumulators = 64 VGPRs).
// MT=4 (64x64) was tried and spilled accumulators to scratch inside the hot
// loop (allocator exceeded the 256-VGPR window with the scheduler's B-frag
// pipelining); 32x64 keeps everything register-resident at ~43 flop/byte of
// L2 traffic. B operands are stored pre-transposed (BT[n][k]) so both
// fragments load as contiguous 16B-aligned b128s matching the CDNA5 16-bit
// WMMA lane layouts.
// ---------------------------------------------------------------------------

#define BATCH   8
#define SEQ     2048
#define EMBED   2048
#define HEADS   16
#define HD      128
#define BH      (BATCH * HEADS)   // 128
#define BS      (BATCH * SEQ)     // 16384
#define EPSV    1e-6f

typedef __attribute__((ext_vector_type(16))) __bf16 v16bf;
typedef __attribute__((ext_vector_type(8)))  float  v8f;

union FragBF {
    unsigned int u[8];
    v16bf        v;
};

__device__ __forceinline__ unsigned short f2bf(float f) {
    unsigned int x = __builtin_bit_cast(unsigned int, f);
    x += 0x7FFFu + ((x >> 16) & 1u);          // round-to-nearest-even
    return (unsigned short)(x >> 16);
}
__device__ __forceinline__ float bf2f(unsigned short u) {
    return __builtin_bit_cast(float, ((unsigned int)u) << 16);
}
__device__ __forceinline__ float phi_elu1(float x) {
    return (x >= 0.0f) ? (x + 1.0f) : __expf(x);   // elu(x)+1
}

// ---------------------------------------------------------------------------
// Wave-level GEMM core: C[32 x 64] += A[32 x K] * B[K x 64]
//   A  : row-major bf16, leading dim lda (elements), tile starts at row m0
//   BT : B stored transposed, BT[n][k], leading dim ldbt, tile at col n0
// acc[i][t] covers rows m0+i*16..+15, cols n0+t*16..+15.
// CDNA5 16-bit A layout: lane<16 -> row=lane, K pairs {0,1..6,7},{16..23};
// lane>=16 same rows, K ranges shifted by +8. B layout: same K order as A,
// column = lane&15, packed pairs per VGPR.
// ---------------------------------------------------------------------------
#define MT 2
__device__ __forceinline__ void wmma_gemm_tile(
    const unsigned short* __restrict__ A, int lda, int m0,
    const unsigned short* __restrict__ BT, int ldbt, int n0,
    int K, v8f acc[MT][4])
{
    const unsigned lane = threadIdx.x & 31u;
    const unsigned half = lane >> 4;       // 0 or 1
    const unsigned mr   = lane & 15u;      // row within subtile (A), col (B)

    const unsigned short* arow0 = A + (size_t)(m0 + mr) * (size_t)lda + half * 8;
    const unsigned short* brow  = BT + (size_t)(n0 + mr) * (size_t)ldbt + half * 8;

    for (int k0 = 0; k0 < K; k0 += 32) {
        FragBF a[MT];
#pragma unroll
        for (int i = 0; i < MT; ++i) {
            const unsigned short* ap = arow0 + (size_t)(i * 16) * (size_t)lda + k0;
#pragma unroll
            for (int p = 0; p < 4; ++p)
                a[i].u[p] = *(const unsigned int*)(ap + 2 * p);
#pragma unroll
            for (int p = 0; p < 4; ++p)
                a[i].u[4 + p] = *(const unsigned int*)(ap + 16 + 2 * p);
        }

#pragma unroll
        for (int t = 0; t < 4; ++t) {
            FragBF b;
            const unsigned short* bp = brow + (size_t)(t * 16) * (size_t)ldbt + k0;
#pragma unroll
            for (int p = 0; p < 8; ++p)
                b.u[p] = *(const unsigned int*)(bp + 2 * p + (p >= 4 ? 8 : 0));

#pragma unroll
            for (int i = 0; i < MT; ++i)
                acc[i][t] = __builtin_amdgcn_wmma_f32_16x16x32_bf16(
                    false, a[i].v, false, b.v, (short)0, acc[i][t], false, false);
        }
    }
}

// C layout for f32 16x16 D: VGPR r -> M = r + 8*(lane>>4), N = lane&15.
#define C_ROW(r)  ((int)((r) + 8 * (threadIdx.x >> 4 & 1)))
#define C_COL()   ((int)(threadIdx.x & 15))

// ---------------------------------------------------------------------------
// Packing kernels (f32 -> bf16, with layout changes)
// ---------------------------------------------------------------------------

// inputs (B,S,E) f32 -> Xbf (BH, S, HD) bf16 row-major (head-split)
__global__ void pack_x_kernel(const float* __restrict__ in,
                              unsigned short* __restrict__ Xbf)
{
    const long long n = (long long)BH * SEQ * HD;
    for (long long o = (long long)blockIdx.x * blockDim.x + threadIdx.x;
         o < n; o += (long long)gridDim.x * blockDim.x) {
        int j = (int)(o % HD);
        long long t = o / HD;
        int s = (int)(t % SEQ);
        int g = (int)(t / SEQ);
        int h = g & (HEADS - 1);
        int b = g >> 4;
        Xbf[o] = f2bf(in[((long long)(b * SEQ + s)) * EMBED + h * HD + j]);
    }
}

// W (Hn, dm, dm) f32 -> WT (Hn, dm, dm) bf16 with WT[h][e][j] = W[h][j][e]
__global__ void pack_wT_kernel(const float* __restrict__ W,
                               unsigned short* __restrict__ WT,
                               int Hn, int dm)
{
    const long long n = (long long)Hn * dm * dm;
    for (long long o = (long long)blockIdx.x * blockDim.x + threadIdx.x;
         o < n; o += (long long)gridDim.x * blockDim.x) {
        int j = (int)(o % dm);
        long long t = o / dm;
        int e = (int)(t % dm);
        int h = (int)(t / dm);
        WT[o] = f2bf(W[((long long)h * dm + j) * dm + e]);
    }
}

// ---------------------------------------------------------------------------
// Q/K/V projection (batched GEMM, 3 weight sets) with phi epilogue.
//   pq  (BH, S, HD)  bf16 row-major            (A for "num" GEMM)
//   pkT (BH, HD, S)  bf16 (feature-major)      (A for "kv" GEMM)
//   vT  (BH, HD, S)  bf16 (feature-major)      (BT for "kv" GEMM)
// ---------------------------------------------------------------------------
__global__ void qkv_proj_kernel(const unsigned short* __restrict__ Xbf,
                                const unsigned short* __restrict__ WqT,
                                const unsigned short* __restrict__ WkT,
                                const unsigned short* __restrict__ WvT,
                                unsigned short* __restrict__ pq,
                                unsigned short* __restrict__ pkT,
                                unsigned short* __restrict__ vT)
{
    // waves: mat(3) x g(128) x mtile(SEQ/32=64) x ngroup(HD/64=2)
    unsigned t = blockIdx.x * (blockDim.x >> 5) + (threadIdx.x >> 5);
    const unsigned total = 3u * BH * (SEQ / 32) * (HD / 64);
    if (t >= total) return;
    const int ncg = t & 1;  t >>= 1;
    const int mt  = t & 63; t >>= 6;
    const int g   = t & 127; t >>= 7;
    const int mat = t;                 // 0=Q, 1=K, 2=V
    const int h   = g & (HEADS - 1);
    const int m0  = mt * 32;
    const int n0  = ncg * 64;

    const unsigned short* A  = Xbf + (size_t)g * SEQ * HD;
    const unsigned short* BT =
        (mat == 0 ? WqT : (mat == 1 ? WkT : WvT)) + (size_t)h * HD * HD;

    v8f acc[MT][4] = {};
    wmma_gemm_tile(A, HD, m0, BT, HD, n0, HD, acc);

    const int col = C_COL();
#pragma unroll
    for (int i = 0; i < MT; ++i) {
#pragma unroll
        for (int tt = 0; tt < 4; ++tt) {
#pragma unroll
            for (int r = 0; r < 8; ++r) {
                const int m = m0 + i * 16 + C_ROW(r);
                const int n = n0 + tt * 16 + col;
                float x = acc[i][tt][r];
                if (mat == 0) {
                    pq[((size_t)g * SEQ + m) * HD + n] = f2bf(phi_elu1(x));
                } else if (mat == 1) {
                    pkT[((size_t)g * HD + n) * SEQ + m] = f2bf(phi_elu1(x));
                } else {
                    vT[((size_t)g * HD + n) * SEQ + m] = f2bf(x);
                }
            }
        }
    }
}

// ksum[g][e] = sum_s pkT[g][e][s]   (one wave per (g,e) row; wave32 shuffle)
__global__ void ksum_kernel(const unsigned short* __restrict__ pkT,
                            float* __restrict__ ksum)
{
    const unsigned row = blockIdx.x * (blockDim.x >> 5) + (threadIdx.x >> 5);
    if (row >= (unsigned)(BH * HD)) return;
    const unsigned lane = threadIdx.x & 31u;
    const unsigned short* p = pkT + (size_t)row * SEQ;
    float sum = 0.0f;
    for (int s = lane; s < SEQ; s += 32) sum += bf2f(p[s]);
#pragma unroll
    for (int off = 16; off > 0; off >>= 1) sum += __shfl_down(sum, off, 32);
    if (lane == 0) ksum[row] = sum;
}

// kvT[g][e2][e1] = sum_s pkT[g][e1][s] * vT[g][e2][s]   (batched 128x128xK2048)
__global__ void kv_kernel(const unsigned short* __restrict__ pkT,
                          const unsigned short* __restrict__ vT,
                          unsigned short* __restrict__ kvT)
{
    unsigned t = blockIdx.x * (blockDim.x >> 5) + (threadIdx.x >> 5);
    const unsigned total = BH * (HD / 32) * (HD / 64);
    if (t >= total) return;
    const int ncg = t & 1; t >>= 1;
    const int mt  = t & 3; t >>= 2;
    const int g   = t;
    const int m0  = mt * 32;
    const int n0  = ncg * 64;

    const unsigned short* A  = pkT + (size_t)g * HD * SEQ;   // (HD x SEQ)
    const unsigned short* BT = vT  + (size_t)g * HD * SEQ;   // BT[n][k]=v[k][n]

    v8f acc[MT][4] = {};
    wmma_gemm_tile(A, SEQ, m0, BT, SEQ, n0, SEQ, acc);

    const int col = C_COL();
#pragma unroll
    for (int i = 0; i < MT; ++i) {
#pragma unroll
        for (int tt = 0; tt < 4; ++tt) {
#pragma unroll
            for (int r = 0; r < 8; ++r) {
                const int m = m0 + i * 16 + C_ROW(r);      // e1 (num contraction)
                const int n = n0 + tt * 16 + col;          // e2 (output feature)
                kvT[((size_t)g * HD + n) * HD + m] = f2bf(acc[i][tt][r]);
            }
        }
    }
}

// den[g][s] = sum_e pq[g][s][e] * ksum[g][e]
__global__ void den_kernel(const unsigned short* __restrict__ pq,
                           const float* __restrict__ ksum,
                           float* __restrict__ den)
{
    const unsigned i = blockIdx.x * blockDim.x + threadIdx.x;
    if (i >= (unsigned)(BH * SEQ)) return;
    const int g = i / SEQ;
    const unsigned short* p = pq + (size_t)i * HD;
    const float* kp = ksum + (size_t)g * HD;
    float sum = 0.0f;
#pragma unroll 4
    for (int e = 0; e < HD; ++e) sum += bf2f(p[e]) * kp[e];
    den[i] = sum;
}

// num = pq @ kv, fused /(den+eps), head-recombine -> combined (B,S,E) bf16
__global__ void num_kernel(const unsigned short* __restrict__ pq,
                           const unsigned short* __restrict__ kvT,
                           const float* __restrict__ den,
                           unsigned short* __restrict__ combined)
{
    unsigned t = blockIdx.x * (blockDim.x >> 5) + (threadIdx.x >> 5);
    const unsigned total = BH * (SEQ / 32) * (HD / 64);
    if (t >= total) return;
    const int ncg = t & 1;  t >>= 1;
    const int mt  = t & 63; t >>= 6;
    const int g   = t;
    const int h   = g & (HEADS - 1);
    const int b   = g >> 4;
    const int m0  = mt * 32;
    const int n0  = ncg * 64;

    const unsigned short* A  = pq  + (size_t)g * SEQ * HD;
    const unsigned short* BT = kvT + (size_t)g * HD * HD;    // BT[n][k]=kv[k][n]

    v8f acc[MT][4] = {};
    wmma_gemm_tile(A, HD, m0, BT, HD, n0, HD, acc);

    const int col = C_COL();
#pragma unroll
    for (int i = 0; i < MT; ++i) {
#pragma unroll
        for (int tt = 0; tt < 4; ++tt) {
#pragma unroll
            for (int r = 0; r < 8; ++r) {
                const int s = m0 + i * 16 + C_ROW(r);
                const int n = n0 + tt * 16 + col;
                const float d = den[(size_t)g * SEQ + s] + EPSV;
                combined[((size_t)(b * SEQ + s)) * EMBED + h * HD + n] =
                    f2bf(acc[i][tt][r] / d);
            }
        }
    }
}

// out = combined @ Wo + bo  (16384 x 2048 x 2048, f32 output)
__global__ void outproj_kernel(const unsigned short* __restrict__ combined,
                               const unsigned short* __restrict__ WoT,
                               const float* __restrict__ bo,
                               float* __restrict__ out)
{
    unsigned t = blockIdx.x * (blockDim.x >> 5) + (threadIdx.x >> 5);
    const unsigned total = (BS / 32) * (EMBED / 64);
    if (t >= total) return;
    const int ncg = t & 31; t >>= 5;
    const int mt  = t;
    const int m0  = mt * 32;
    const int n0  = ncg * 64;

    v8f acc[MT][4] = {};
    wmma_gemm_tile(combined, EMBED, m0, WoT, EMBED, n0, EMBED, acc);

    const int col = C_COL();
#pragma unroll
    for (int i = 0; i < MT; ++i) {
#pragma unroll
        for (int tt = 0; tt < 4; ++tt) {
#pragma unroll
            for (int r = 0; r < 8; ++r) {
                const int m = m0 + i * 16 + C_ROW(r);
                const int n = n0 + tt * 16 + col;
                out[(size_t)m * EMBED + n] = acc[i][tt][r] + bo[n];
            }
        }
    }
}

// ---------------------------------------------------------------------------
// Host launcher
// ---------------------------------------------------------------------------
extern "C" void kernel_launch(void* const* d_in, const int* in_sizes, int n_in,
                              void* d_out, int out_size, void* d_ws, size_t ws_size,
                              hipStream_t stream)
{
    const float* in  = (const float*)d_in[0];   // (B,S,E)
    const float* Wq  = (const float*)d_in[1];   // (H,d,d)
    const float* Wk  = (const float*)d_in[2];
    const float* Wv  = (const float*)d_in[3];
    const float* Wo  = (const float*)d_in[4];   // (E,E)
    const float* bo  = (const float*)d_in[5];   // (E)
    float* out = (float*)d_out;

    // workspace carve-up (256B aligned); total ~285 MB
    size_t off = 0;
    auto carve = [&](size_t bytes) {
        size_t o = off;
        off = (off + bytes + 255) & ~(size_t)255;
        return o;
    };
    char* ws = (char*)d_ws;
    const size_t xbf_o  = carve((size_t)BH * SEQ * HD * 2);   // 64 MB (reused as `combined`)
    const size_t wqT_o  = carve((size_t)HEADS * HD * HD * 2);
    const size_t wkT_o  = carve((size_t)HEADS * HD * HD * 2);
    const size_t wvT_o  = carve((size_t)HEADS * HD * HD * 2);
    const size_t woT_o  = carve((size_t)EMBED * EMBED * 2);   // 8 MB
    const size_t pq_o   = carve((size_t)BH * SEQ * HD * 2);   // 64 MB
    const size_t pkT_o  = carve((size_t)BH * HD * SEQ * 2);   // 64 MB
    const size_t vT_o   = carve((size_t)BH * HD * SEQ * 2);   // 64 MB
    const size_t kvT_o  = carve((size_t)BH * HD * HD * 2);    // 4 MB
    const size_t ksum_o = carve((size_t)BH * HD * 4);
    const size_t den_o  = carve((size_t)BH * SEQ * 4);

    unsigned short* Xbf      = (unsigned short*)(ws + xbf_o);
    unsigned short* WqT      = (unsigned short*)(ws + wqT_o);
    unsigned short* WkT      = (unsigned short*)(ws + wkT_o);
    unsigned short* WvT      = (unsigned short*)(ws + wvT_o);
    unsigned short* WoT      = (unsigned short*)(ws + woT_o);
    unsigned short* pq       = (unsigned short*)(ws + pq_o);
    unsigned short* pkT      = (unsigned short*)(ws + pkT_o);
    unsigned short* vT       = (unsigned short*)(ws + vT_o);
    unsigned short* kvT      = (unsigned short*)(ws + kvT_o);
    float*          ksum     = (float*)(ws + ksum_o);
    float*          den      = (float*)(ws + den_o);
    unsigned short* combined = Xbf;   // Xbf dead after qkv_proj -> alias

    const int T = 256;  // 8 waves/block

    // 1. pack/convert
    pack_x_kernel<<<8192, T, 0, stream>>>(in, Xbf);
    pack_wT_kernel<<<1024, T, 0, stream>>>(Wq, WqT, HEADS, HD);
    pack_wT_kernel<<<1024, T, 0, stream>>>(Wk, WkT, HEADS, HD);
    pack_wT_kernel<<<1024, T, 0, stream>>>(Wv, WvT, HEADS, HD);
    pack_wT_kernel<<<8192, T, 0, stream>>>(Wo, WoT, 1, EMBED);

    // 2. fused Q/K/V projection + phi   (49152 waves)
    qkv_proj_kernel<<<(3 * BH * (SEQ / 32) * (HD / 64)) / 8, T, 0, stream>>>(
        Xbf, WqT, WkT, WvT, pq, pkT, vT);

    // 3. ksum reduction (16384 waves)
    ksum_kernel<<<(BH * HD) / 8, T, 0, stream>>>(pkT, ksum);

    // 4. kv = phi(k)^T v   (1024 waves)
    kv_kernel<<<(BH * (HD / 32) * (HD / 64)) / 8, T, 0, stream>>>(pkT, vT, kvT);

    // 5. den = phi(q) . ksum
    den_kernel<<<(BH * SEQ) / T, T, 0, stream>>>(pq, ksum, den);

    // 6. num = phi(q) @ kv, fused /(den+eps), head recombine (16384 waves)
    num_kernel<<<(BH * (SEQ / 32) * (HD / 64)) / 8, T, 0, stream>>>(
        pq, kvT, den, combined);

    // 7. out = combined @ Wo + bo   (16384 waves)
    outproj_kernel<<<((BS / 32) * (EMBED / 64)) / 8, T, 0, stream>>>(
        combined, WoT, bo, out);

    (void)in_sizes; (void)n_in; (void)out_size; (void)ws_size;
}
